// MultiHeadAttention_33328946217123
// MI455X (gfx1250) — matrix operand order repaired and checked by reading, not compiled, not run on hardware
//
#include <hip/hip_runtime.h>
#include <hip/hip_bf16.h>

// CDNA5 / gfx1250: wave32, WMMA 16x16x32 bf16 (f32 accum), TDM async tensor loads.

typedef __bf16 bf16_t;
typedef __attribute__((ext_vector_type(4)))  bf16_t v4bf;
typedef __attribute__((ext_vector_type(8)))  bf16_t v8bf;
typedef __attribute__((ext_vector_type(16))) bf16_t v16bf;
typedef __attribute__((ext_vector_type(8)))  float  v8f;
typedef __attribute__((ext_vector_type(4)))  unsigned int v4u;
typedef __attribute__((ext_vector_type(8)))  int v8i;
typedef __attribute__((ext_vector_type(4)))  int v4i;

#define BB   2
#define TT   2048
#define DD   2048
#define NH   16
#define HD   128
#define LQKV 6144   // fused QKV row stride = 3*DD

static __device__ inline v16bf cat8(v8bf lo, v8bf hi) {
  v16bf r;
#pragma unroll
  for (int i = 0; i < 8; ++i) { r[i] = lo[i]; r[i + 8] = hi[i]; }
  return r;
}

// A-fragment (16x32 bf16): lane m -> row m; lanes 0-15 hold K {0..7,16..23},
// lanes 16-31 hold K {8..15,24..31}. Caller pre-offsets p by khalf.
static __device__ inline v16bf load_fragA(const bf16_t* p) {
  v8bf lo = *(const v8bf*)p;          // K khalf+0..7
  v8bf hi = *(const v8bf*)(p + 16);   // K khalf+16..23
  return cat8(lo, hi);
}
// B-fragment (32x16 bf16): lane n -> col n; K {khalf+0..7, khalf+16..23}.
static __device__ inline v16bf load_fragB(const bf16_t* p) {
  v8bf lo = *(const v8bf*)p;
  v8bf hi = *(const v8bf*)(p + 16);
  return cat8(lo, hi);
}

static __device__ inline v8f wmma_bf16(v16bf a, v16bf b, v8f c) {
  return __builtin_amdgcn_wmma_f32_16x16x32_bf16(
      false, a, false, b, (short)0, c, false, false);
}

// ---- Tensor Data Mover: 2D tile (tile_rows x 32 bf16) global -> LDS, with
// LDS padding of 4 DWORDs after every 16 DWORDs (row stride 64B -> 80B,
// matching the 40-element LDS row pitch the WMMA fragment loads use).
static __device__ inline void tdm_load_tile32(unsigned lds_byte_addr,
                                              const bf16_t* gptr,
                                              unsigned tile_rows,
                                              unsigned row_stride_elems) {
  const unsigned long long ga = (unsigned long long)(uintptr_t)gptr;
  v4u g0;
  g0[0] = 1u;                                   // count=1 (valid), user mode
  g0[1] = lds_byte_addr;                        // LDS destination (bytes)
  g0[2] = (unsigned)ga;                         // global_addr[31:0]
  g0[3] = (unsigned)(ga >> 32) | (2u << 30);    // global_addr[56:32] | type=2
  v8i g1;
  g1[0] = (int)((1u << 16)      // data_size = 2 bytes
              | (1u << 20)      // pad_enable
              | (3u << 22)      // pad_interval code 3 = 16 DWORDs (one 64B row)
              | (3u << 25));    // pad_amount  code 3 = 4 DWORDs (16B)
  g1[1] = (int)((32u & 0xFFFFu) << 16);                 // tensor_dim0 = 32
  g1[2] = (int)((tile_rows & 0xFFFFu) << 16);           // tensor_dim1 = rows
  g1[3] = (int)(32u << 16);                             // tile_dim0 = 32
  g1[4] = (int)tile_rows;                               // tile_dim1 (tile_dim2=0)
  g1[5] = (int)row_stride_elems;                        // tensor_dim0_stride lo
  g1[6] = 0;
  g1[7] = 0;
  const v4i gz4 = {0, 0, 0, 0};
  const v8i gz8 = {0, 0, 0, 0, 0, 0, 0, 0};
  // clang-23 / therock form: (g0, g1, g2, g3, g4, cpol)
  __builtin_amdgcn_tensor_load_to_lds(g0, g1, gz4, gz4, gz8, 0);
}

// Interleaved-pair RoPE on 8 contiguous bf16 lanes (dbase even).
static __device__ inline void rope8(v8bf& v, float pos, int dbase) {
#pragma unroll
  for (int j = 0; j < 4; ++j) {
    const float fi = (float)((dbase >> 1) + j);          // pair index i
    const float invf = __expf(fi * (-9.210340371976184f / 64.0f)); // theta^(-2i/128)
    float s, c;
    __sincosf(pos * invf, &s, &c);
    const float x1 = (float)v[2 * j];
    const float x2 = (float)v[2 * j + 1];
    v[2 * j]     = (bf16_t)(x1 * c - x2 * s);
    v[2 * j + 1] = (bf16_t)(x1 * s + x2 * c);
  }
}

// ---------------------------------------------------------------- convert x
__global__ __launch_bounds__(256) void k_convert_x(
    const float* __restrict__ x, bf16_t* __restrict__ xb, int n4) {
  const int i = blockIdx.x * 256 + threadIdx.x;
  if (i >= n4) return;
  const float4 f = ((const float4*)x)[i];
  v4bf o;
  o[0] = (bf16_t)f.x; o[1] = (bf16_t)f.y; o[2] = (bf16_t)f.z; o[3] = (bf16_t)f.w;
  ((v4bf*)xb)[i] = o;
}

// -------------------------------------------- transpose + convert weights
__global__ __launch_bounds__(256) void k_convert_w(
    const float* __restrict__ Wq, const float* __restrict__ Wk,
    const float* __restrict__ Wv, const float* __restrict__ Wo,
    bf16_t* __restrict__ wt_qkv, bf16_t* __restrict__ wt_o) {
  __shared__ float tile[32][33];
  const int widx = blockIdx.z;
  const float* W = (widx == 0) ? Wq : (widx == 1) ? Wk : (widx == 2) ? Wv : Wo;
  bf16_t* out = (widx < 3) ? (wt_qkv + (size_t)widx * DD * DD) : wt_o;
  const int tx = threadIdx.x, ty = threadIdx.y;
  const int n0 = blockIdx.x * 32, k0 = blockIdx.y * 32;
#pragma unroll
  for (int i = 0; i < 4; ++i)
    tile[ty + 8 * i][tx] = W[(size_t)(k0 + ty + 8 * i) * DD + n0 + tx];
  __syncthreads();
#pragma unroll
  for (int i = 0; i < 4; ++i)
    out[(size_t)(n0 + ty + 8 * i) * DD + k0 + tx] = (bf16_t)tile[tx][ty + 8 * i];
}

// ------------------------------------------- TDM-staged WMMA GEMM (bf16)
// C[M][ldc] = A[M][K](bf16) * Bt[N][K]^T(bf16).  Block tile 128x256,
// 8 waves in a 2(M) x 4(N) grid, wave tile 64x64 (4x4 WMMA subtiles).
// K stepped by 32; tiles double-buffered in LDS, filled by the Tensor Data
// Mover (wave 0 issues, all waves sync through s_wait_tensorcnt + barrier).
template <bool OUT_BF16>
__global__ __launch_bounds__(256) void k_gemm(
    const bf16_t* __restrict__ A, const bf16_t* __restrict__ Bt,
    void* __restrict__ Cout, int K, int ldc) {
  __shared__ __attribute__((aligned(16))) bf16_t sA[2][128 * 40];
  __shared__ __attribute__((aligned(16))) bf16_t sB[2][256 * 40];
  const int tid = threadIdx.x;
  const int wid = tid >> 5, lane = tid & 31;
  const size_t m0 = (size_t)blockIdx.y * 128;
  const size_t n0 = (size_t)blockIdx.x * 256;
  const int wm = (wid & 1) * 64;        // 0 or 64
  const int wn = (wid >> 1) * 64;       // 0,64,128,192
  const int lf = lane & 15;
  const int khalfA = (lane >> 4) * 8;
  const int koffB  = (lane >> 4) * 8;

  v8f acc[4][4];
  const v8f z = {0.f, 0.f, 0.f, 0.f, 0.f, 0.f, 0.f, 0.f};
#pragma unroll
  for (int i = 0; i < 4; ++i)
#pragma unroll
    for (int j = 0; j < 4; ++j) acc[i][j] = z;

  const unsigned ldsA0 = (unsigned)(uintptr_t)&sA[0][0];
  const unsigned ldsA1 = (unsigned)(uintptr_t)&sA[1][0];
  const unsigned ldsB0 = (unsigned)(uintptr_t)&sB[0][0];
  const unsigned ldsB1 = (unsigned)(uintptr_t)&sB[1][0];

  const int S = K >> 5;                 // number of 32-deep K steps
  if (wid == 0) {                       // prefetch steps 0 and 1
    tdm_load_tile32(ldsA0, &A[m0 * (size_t)K + 0],  128, (unsigned)K);
    tdm_load_tile32(ldsB0, &Bt[n0 * (size_t)K + 0], 256, (unsigned)K);
    tdm_load_tile32(ldsA1, &A[m0 * (size_t)K + 32],  128, (unsigned)K);
    tdm_load_tile32(ldsB1, &Bt[n0 * (size_t)K + 32], 256, (unsigned)K);
  }

  for (int s = 0; s < S; ++s) {
    if (wid == 0) {                     // oldest 2 TDM ops (this step) done
      if (s + 1 < S) __builtin_amdgcn_s_wait_tensorcnt((short)2);
      else           __builtin_amdgcn_s_wait_tensorcnt((short)0);
    }
    __syncthreads();
    const bf16_t* bA = &sA[s & 1][0];
    const bf16_t* bB = &sB[s & 1][0];
    v16bf af[4], bfr[4];
#pragma unroll
    for (int i = 0; i < 4; ++i)
      af[i] = load_fragA(&bA[(wm + i * 16 + lf) * 40 + khalfA]);
#pragma unroll
    for (int j = 0; j < 4; ++j)
      bfr[j] = load_fragB(&bB[(wn + j * 16 + lf) * 40 + koffB]);
#pragma unroll
    for (int i = 0; i < 4; ++i)
#pragma unroll
      for (int j = 0; j < 4; ++j)
        acc[i][j] = wmma_bf16(af[i], bfr[j], acc[i][j]);
    __syncthreads();                    // everyone done reading buffer s&1
    if (wid == 0 && s + 2 < S) {        // refill it for step s+2
      const size_t k0 = (size_t)(s + 2) * 32;
      tdm_load_tile32((s & 1) ? ldsA1 : ldsA0, &A[m0 * (size_t)K + k0],  128, (unsigned)K);
      tdm_load_tile32((s & 1) ? ldsB1 : ldsB0, &Bt[n0 * (size_t)K + k0], 256, (unsigned)K);
    }
  }

  const int nout = lane & 15, mb = (lane >> 4) * 8;
#pragma unroll
  for (int i = 0; i < 4; ++i)
#pragma unroll
    for (int j = 0; j < 4; ++j)
#pragma unroll
      for (int r = 0; r < 8; ++r) {
        const size_t off =
            (m0 + wm + i * 16 + mb + r) * (size_t)ldc + n0 + wn + j * 16 + nout;
        if constexpr (OUT_BF16) ((bf16_t*)Cout)[off] = (bf16_t)acc[i][j][r];
        else                    ((float*)Cout)[off]  = acc[i][j][r];
      }
}

// ------------------------------------------ V -> Vt[bh][d][T] (bf16 copy)
__global__ __launch_bounds__(256) void k_transpose_v(
    const bf16_t* __restrict__ qkv, bf16_t* __restrict__ vt) {
  __shared__ bf16_t tile[32][33];
  const int bh = blockIdx.z, b = bh >> 4, h = bh & 15;
  const int t0 = blockIdx.x * 32, d0 = blockIdx.y * 32;
  const int tx = threadIdx.x, ty = threadIdx.y;
#pragma unroll
  for (int i = 0; i < 4; ++i)
    tile[ty + 8 * i][tx] =
        qkv[(size_t)(b * TT + t0 + ty + 8 * i) * LQKV + 2 * DD + h * HD + d0 + tx];
  __syncthreads();
#pragma unroll
  for (int i = 0; i < 4; ++i)
    vt[((size_t)bh * HD + d0 + ty + 8 * i) * TT + t0 + tx] = tile[tx][ty + 8 * i];
}

// -------------------------------------------------- flash attention (bf16)
// One wave handles (bh, 16-query tile); RoPE applied in-register on Q/K.
__global__ __launch_bounds__(256) void k_attn(
    const bf16_t* __restrict__ qkv, const bf16_t* __restrict__ vt,
    bf16_t* __restrict__ ctx) {
  __shared__ __attribute__((aligned(16))) bf16_t psh[8 * 16 * 40];
  const int tid = threadIdx.x, wid = tid >> 5, lane = tid & 31;
  const int task = blockIdx.x * 8 + wid;       // 0..4095
  const int bh = task >> 7, qt = task & 127;
  const int b = bh >> 4, h = bh & 15;
  const int q0 = qt * 16;
  const int lf = lane & 15;
  const int khalf = (lane >> 4) * 8;
  const int koff  = (lane >> 4) * 8;
  const int mb    = (lane >> 4) * 8;

  const bf16_t* qbase = qkv + (size_t)(b * TT + q0) * LQKV + h * HD;
  const bf16_t* kmat  = qkv + (size_t)b * TT * LQKV + DD + h * HD;
  const bf16_t* vbase = vt + (size_t)bh * HD * TT;
  bf16_t* ps = psh + wid * (16 * 40);

  // Q fragments (d = 128 -> 4 A-frags), RoPE once.
  v16bf qf[4];
#pragma unroll
  for (int kk = 0; kk < 4; ++kk) {
    const int dk = kk * 32;
    const bf16_t* p = qbase + (size_t)lf * LQKV + dk + khalf;
    v8bf lo = *(const v8bf*)p;
    v8bf hi = *(const v8bf*)(p + 16);
    rope8(lo, (float)(q0 + lf), dk + khalf);
    rope8(hi, (float)(q0 + lf), dk + khalf + 16);
    qf[kk] = cat8(lo, hi);
  }

  const v8f z = {0.f, 0.f, 0.f, 0.f, 0.f, 0.f, 0.f, 0.f};
  v8f o[8];
  float mr[8], lr[8];
#pragma unroll
  for (int c = 0; c < 8; ++c) o[c] = z;
#pragma unroll
  for (int r = 0; r < 8; ++r) { mr[r] = -3.0e38f; lr[r] = 0.f; }

  const float scale = 0.08838834764831845f;    // 1/sqrt(128)
  const int ktmax = (q0 + 15) >> 5;
  for (int kt = 0; kt <= ktmax; ++kt) {
    const int j0 = kt * 32;
    v8f s0 = z, s1 = z;
#pragma unroll
    for (int kk = 0; kk < 4; ++kk) {           // S = Q * K^T over d
      const int dk = kk * 32;
      {
        const bf16_t* p = kmat + (size_t)(j0 + lf) * LQKV + dk + koff;
        v8bf lo = *(const v8bf*)p;
        v8bf hi = *(const v8bf*)(p + 16);
        rope8(lo, (float)(j0 + lf), dk + koff);
        rope8(hi, (float)(j0 + lf), dk + koff + 16);
        s0 = wmma_bf16(qf[kk], cat8(lo, hi), s0);
      }
      {
        const bf16_t* p = kmat + (size_t)(j0 + 16 + lf) * LQKV + dk + koff;
        v8bf lo = *(const v8bf*)p;
        v8bf hi = *(const v8bf*)(p + 16);
        rope8(lo, (float)(j0 + 16 + lf), dk + koff);
        rope8(hi, (float)(j0 + 16 + lf), dk + koff + 16);
        s1 = wmma_bf16(qf[kk], cat8(lo, hi), s1);
      }
    }

    const bool diag = (j0 + 31 > q0);
#pragma unroll
    for (int r = 0; r < 8; ++r) {
      float a0 = s0[r] * scale;
      float a1 = s1[r] * scale;
      if (diag) {
        const int q = q0 + mb + r;
        if (j0 + lf > q)      a0 = -3.0e38f;
        if (j0 + 16 + lf > q) a1 = -3.0e38f;
      }
      float rm = fmaxf(a0, a1);                // row-max across 16 lanes
      rm = fmaxf(rm, __shfl_xor(rm, 1, 32));
      rm = fmaxf(rm, __shfl_xor(rm, 2, 32));
      rm = fmaxf(rm, __shfl_xor(rm, 4, 32));
      rm = fmaxf(rm, __shfl_xor(rm, 8, 32));
      const float mnew = fmaxf(mr[r], rm);
      const float alpha = __expf(mr[r] - mnew);
      mr[r] = mnew;
      const float e0 = __expf(a0 - mnew);
      const float e1 = __expf(a1 - mnew);
      float rs = e0 + e1;
      rs += __shfl_xor(rs, 1, 32);
      rs += __shfl_xor(rs, 2, 32);
      rs += __shfl_xor(rs, 4, 32);
      rs += __shfl_xor(rs, 8, 32);
      lr[r] = lr[r] * alpha + rs;
#pragma unroll
      for (int c = 0; c < 8; ++c) o[c][r] *= alpha;
      ps[(mb + r) * 40 + lf]      = (bf16_t)e0;   // P tile, row-major [16][32]
      ps[(mb + r) * 40 + 16 + lf] = (bf16_t)e1;
    }

    // C-layout -> A-layout via wave-private LDS (DS ops in-order per wave).
    const v16bf pf = load_fragA(ps + (size_t)lf * 40 + khalf);
#pragma unroll
    for (int c = 0; c < 8; ++c) {               // O += P * V
      const bf16_t* vp = vbase + (size_t)(c * 16 + lf) * TT + j0 + koff;
      v8bf lo = *(const v8bf*)vp;
      v8bf hi = *(const v8bf*)(vp + 16);
      o[c] = wmma_bf16(pf, cat8(lo, hi), o[c]);
    }
  }

#pragma unroll
  for (int r = 0; r < 8; ++r) lr[r] = 1.0f / lr[r];
  bf16_t* cbase = ctx + (size_t)(b * TT + q0) * DD + h * HD;
#pragma unroll
  for (int c = 0; c < 8; ++c)
#pragma unroll
    for (int r = 0; r < 8; ++r)
      cbase[(size_t)(mb + r) * DD + c * 16 + lf] = (bf16_t)(o[c][r] * lr[r]);
}

// --------------------------------------------------------------- launcher
extern "C" void kernel_launch(void* const* d_in, const int* in_sizes, int n_in,
                              void* d_out, int out_size, void* d_ws, size_t ws_size,
                              hipStream_t stream) {
  (void)in_sizes; (void)n_in; (void)out_size; (void)ws_size;
  const float* x  = (const float*)d_in[0];
  const float* Wq = (const float*)d_in[1];
  const float* Wk = (const float*)d_in[2];
  const float* Wv = (const float*)d_in[3];
  const float* Wo = (const float*)d_in[4];
  float* out = (float*)d_out;

  const size_t M = (size_t)BB * TT;            // 4096 rows
  bf16_t* ws      = (bf16_t*)d_ws;
  bf16_t* xb      = ws;                                  //  8M elems
  bf16_t* wt_qkv  = xb + M * DD;                         // 12M elems
  bf16_t* wt_o    = wt_qkv + (size_t)3 * DD * DD;        //  4M elems
  bf16_t* qkv_lin = wt_o + (size_t)DD * DD;              // 24M elems
  bf16_t* vmat_t  = qkv_lin + M * (size_t)LQKV;          //  8M elems
  bf16_t* ctx     = xb;                                  // reuse (xb dead)

  k_convert_x<<<dim3((unsigned)(M * DD / 4 / 256)), dim3(256), 0, stream>>>(
      x, xb, (int)(M * DD / 4));
  k_convert_w<<<dim3(64, 64, 4), dim3(32, 8), 0, stream>>>(
      Wq, Wk, Wv, Wo, wt_qkv, wt_o);
  k_gemm<true><<<dim3(24, 32), dim3(256), 0, stream>>>(    // QKV fused GEMM
      xb, wt_qkv, qkv_lin, DD, LQKV);
  k_transpose_v<<<dim3(64, 4, 32), dim3(32, 8), 0, stream>>>(qkv_lin, vmat_t);
  k_attn<<<dim3(512), dim3(256), 0, stream>>>(qkv_lin, vmat_t, ctx);
  k_gemm<false><<<dim3(8, 32), dim3(256), 0, stream>>>(    // out = ctx @ Wo
      ctx, wt_o, out, DD, DD);
}
